// GeometricAttention_88304527606684
// MI455X (gfx1250) — compile-verified, matrix operand order checked by reading, not run on hardware
//
#include <hip/hip_runtime.h>
#include <hip/hip_bf16.h>

#define BATCH   2
#define SEQ     2048
#define DMODEL  1024
#define NHEADS  16
#define DK      64
#define MROWS   (BATCH * SEQ)     // 4096

typedef __attribute__((ext_vector_type(16))) _Float16 v16h;
typedef __attribute__((ext_vector_type(8)))  _Float16 v8h;
typedef __attribute__((ext_vector_type(8)))  float    v8f;
typedef __attribute__((ext_vector_type(4)))  int      v4i;

#if defined(__gfx1250__) && __has_builtin(__builtin_amdgcn_global_load_async_to_lds_b128) && __has_builtin(__builtin_amdgcn_s_wait_asynccnt)
#define HAS_ASYNC_LDS 1
#else
#define HAS_ASYNC_LDS 0
#endif

// Copy 16 bytes global -> LDS. Uses the CDNA5 async-to-LDS path (ASYNCcnt,
// no VGPR round trip) when the builtin exists; else load+ds_store fallback.
static __device__ __forceinline__ void stage16(_Float16* lds_dst, const _Float16* gsrc) {
#if HAS_ASYNC_LDS
    __builtin_amdgcn_global_load_async_to_lds_b128(
        (__attribute__((address_space(1))) v4i*)gsrc,
        (__attribute__((address_space(3))) v4i*)lds_dst, 0, 0);
#else
    *(v8h*)lds_dst = *(const v8h*)gsrc;
#endif
}

static __device__ __forceinline__ void stage_wait() {
#if HAS_ASYNC_LDS
    __builtin_amdgcn_s_wait_asynccnt(0);
#endif
}

// Load a 16x32 (f16) A/B fragment row for this lane.
// Per CDNA5 ISA layout: lane half 0 holds K = [0..7] and [16..23],
// half 1 holds K = [8..15] and [24..31]. Two aligned 16B loads.
static __device__ __forceinline__ v16h load_frag16(const _Float16* p, int hl) {
    v8h lo = *(const v8h*)(p + hl * 8);
    v8h hi = *(const v8h*)(p + 16 + hl * 8);
    return __builtin_shufflevector(lo, hi, 0, 1, 2, 3, 4, 5, 6, 7,
                                           8, 9, 10, 11, 12, 13, 14, 15);
}

static __device__ __forceinline__ v8f wmma_f16(v16h a, v16h b, v8f c) {
    return __builtin_amdgcn_wmma_f32_16x16x32_f16(
        /*neg_a=*/false, a, /*neg_b=*/false, b,
        /*c_mod=*/(short)0, c, /*reuse_a=*/false, /*reuse_b=*/false);
}

// ---------------------------------------------------------------------------
// f32 -> f16 conversion
// ---------------------------------------------------------------------------
__global__ __launch_bounds__(256)
void cvt_f32_f16(const float* __restrict__ in, _Float16* __restrict__ out, int n) {
    int i = blockIdx.x * blockDim.x + threadIdx.x;
    if (i < n) out[i] = (_Float16)in[i];
}

// ---------------------------------------------------------------------------
// Y = X @ W^T + bias.  X:[MROWS, DMODEL] f16, W:[DMODEL, DMODEL] f16 row-major.
// One wave computes a 32x64 tile (8 accumulators, 8 WMMA per 12 loads).
// K-loop of 32, software-pipelined and unrolled by 2 so the two fragment
// register sets ping-pong (no rotation moves), loads stay in flight across
// the WMMAs of the other set.
// mode 0: f16 store [MROWS, DMODEL]        (Q, K projections)
// mode 1: f16 transposed store Vt[b][c][n] (V projection)
// mode 2: f32 store [MROWS, DMODEL]        (final output projection)
// ---------------------------------------------------------------------------
__global__ __launch_bounds__(128)
void gemm16(const _Float16* __restrict__ X, const _Float16* __restrict__ W,
            const float* __restrict__ bias,
            _Float16* __restrict__ outh, float* __restrict__ outf, int mode) {
    const int lane = threadIdx.x & 31;
    const int wid  = threadIdx.x >> 5;
    const int hl   = lane >> 4;
    const int l16  = lane & 15;

    const int tile  = blockIdx.x * 4 + wid;   // 2048 tiles total
    const int trow  = tile >> 4;              // 128 strips of 32 rows
    const int tcol  = tile & 15;              // 16 groups of 64 cols
    const int rbase = trow * 32;
    const int cbase = tcol * 64;

    const _Float16* arow0 = X + (size_t)(rbase + l16) * DMODEL;
    const _Float16* arow1 = X + (size_t)(rbase + 16 + l16) * DMODEL;
    const _Float16* brow[4];
#pragma unroll
    for (int t = 0; t < 4; ++t)
        brow[t] = W + (size_t)(cbase + t * 16 + l16) * DMODEL;

    v8f acc[2][4];
#pragma unroll
    for (int r = 0; r < 2; ++r)
#pragma unroll
        for (int t = 0; t < 4; ++t) acc[r][t] = v8f{};

    // Set "c" holds fragments for kb; set "n" for kb+32.
    v16h ca0 = load_frag16(arow0, hl);
    v16h ca1 = load_frag16(arow1, hl);
    v16h cb[4];
#pragma unroll
    for (int t = 0; t < 4; ++t) cb[t] = load_frag16(brow[t], hl);

    for (int kb = 0; kb + 64 < DMODEL; kb += 64) {
        // Fragments for kb+32 (issued before consuming set c).
        v16h na0 = load_frag16(arow0 + kb + 32, hl);
        v16h na1 = load_frag16(arow1 + kb + 32, hl);
        v16h nb[4];
#pragma unroll
        for (int t = 0; t < 4; ++t) nb[t] = load_frag16(brow[t] + kb + 32, hl);
        __builtin_prefetch(arow0 + kb + 256, 0, 3);

#pragma unroll
        for (int t = 0; t < 4; ++t) {
            acc[0][t] = wmma_f16(ca0, cb[t], acc[0][t]);
            acc[1][t] = wmma_f16(ca1, cb[t], acc[1][t]);
        }

        // Refill set c with fragments for kb+64 while set n computes.
        ca0 = load_frag16(arow0 + kb + 64, hl);
        ca1 = load_frag16(arow1 + kb + 64, hl);
#pragma unroll
        for (int t = 0; t < 4; ++t) cb[t] = load_frag16(brow[t] + kb + 64, hl);

#pragma unroll
        for (int t = 0; t < 4; ++t) {
            acc[0][t] = wmma_f16(na0, nb[t], acc[0][t]);
            acc[1][t] = wmma_f16(na1, nb[t], acc[1][t]);
        }
    }
    // Epilogue: set c holds kb = DMODEL-64; load and compute the final pair.
    {
        v16h na0 = load_frag16(arow0 + DMODEL - 32, hl);
        v16h na1 = load_frag16(arow1 + DMODEL - 32, hl);
        v16h nb[4];
#pragma unroll
        for (int t = 0; t < 4; ++t)
            nb[t] = load_frag16(brow[t] + DMODEL - 32, hl);
#pragma unroll
        for (int t = 0; t < 4; ++t) {
            acc[0][t] = wmma_f16(ca0, cb[t], acc[0][t]);
            acc[1][t] = wmma_f16(ca1, cb[t], acc[1][t]);
        }
#pragma unroll
        for (int t = 0; t < 4; ++t) {
            acc[0][t] = wmma_f16(na0, nb[t], acc[0][t]);
            acc[1][t] = wmma_f16(na1, nb[t], acc[1][t]);
        }
    }

#pragma unroll
    for (int r = 0; r < 2; ++r) {
#pragma unroll
        for (int t = 0; t < 4; ++t) {
#pragma unroll
            for (int e = 0; e < 8; ++e) {
                const int gr = rbase + r * 16 + e + 8 * hl;  // C/D: row = e + 8*half
                const int gc = cbase + t * 16 + l16;         // col = lane%16
                const float v = acc[r][t][e] + bias[gc];
                if (mode == 2) {
                    outf[(size_t)gr * DMODEL + gc] = v;
                } else if (mode == 1) {
                    const int b_ = gr >> 11;                 // gr / SEQ
                    const int n  = gr & (SEQ - 1);
                    outh[((size_t)b_ * DMODEL + gc) * SEQ + n] = (_Float16)v;
                } else {
                    outh[(size_t)gr * DMODEL + gc] = (_Float16)v;
                }
            }
        }
    }
}

// ---------------------------------------------------------------------------
// Fused flash attention with geometric distance bias.
// One block = 4 waves sharing one (b, h); each wave owns a 16-query tile.
// K/V tiles are double-buffered in LDS: the async copies for key-step j+1
// are issued right after the top barrier of step j and only waited at the
// top of step j+1, so global->LDS latency overlaps WMMA + softmax compute.
// Per step: 4 WMMA (S = Q K^T from LDS), exact distance bias, online
// softmax, P relayout through LDS, 4 WMMA (O += P V from LDS).
// ---------------------------------------------------------------------------
__global__ __launch_bounds__(128)
void attn_fa(const _Float16* __restrict__ Qh, const _Float16* __restrict__ Kh,
             const _Float16* __restrict__ Vt, const float* __restrict__ pos,
             const float* __restrict__ dist_bias, _Float16* __restrict__ O) {
    __shared__ __align__(16) _Float16 Kt[2][32][64];  // [buf][key j][d]
    __shared__ __align__(16) _Float16 Vl[2][64][32];  // [buf][d (c)][key j]
    __shared__ __align__(16) _Float16 Pt[4][16 * 32]; // per-wave P relayout

    const int tid  = threadIdx.x;
    const int lane = tid & 31;
    const int wid  = tid >> 5;
    const int hl   = lane >> 4;
    const int l16  = lane & 15;

    const int bh    = blockIdx.x >> 5;     // 32 (b,h) pairs; 32 blocks each
    const int qgrp  = blockIdx.x & 31;
    const int b     = bh >> 4;
    const int h     = bh & 15;
    const int qbase = (qgrp * 4 + wid) * 16;

    const float ab    = fabsf(dist_bias[h]);
    const float scale = 0.125f;            // 1/sqrt(DK)

    // Query positions for the 8 rows this lane's accumulator elements cover.
    float qx[8], qy[8], qz[8];
#pragma unroll
    for (int e = 0; e < 8; ++e) {
        const int nq = qbase + e + 8 * hl;
        const float* pq = pos + ((size_t)b * SEQ + nq) * 3;
        qx[e] = pq[0]; qy[e] = pq[1]; qz[e] = pq[2];
    }

    // Q fragments: invariant over the key loop -> registers.
    const _Float16* qrow =
        Qh + (size_t)(b * SEQ + qbase + l16) * DMODEL + h * DK;
    const v16h aq0 = load_frag16(qrow, hl);
    const v16h aq1 = load_frag16(qrow + 32, hl);

    v8f o[4] = {v8f{}, v8f{}, v8f{}, v8f{}};
    float m[8], l[8];
#pragma unroll
    for (int e = 0; e < 8; ++e) { m[e] = -3.0e38f; l[e] = 0.0f; }

    // Cooperative staging indices (constant over loop).
    const int krow = tid >> 2, kseg = (tid & 3) * 16;   // K: 32 rows x 128B
    const int vrow = tid >> 1, vseg = (tid & 1) * 16;   // V: 64 rows x 64B

    // Stage one 32-key block (K 32x64 + V 64x32) into LDS buffer `buf`.
    auto stage_tiles = [&](int buf, int jb) {
        const _Float16* ks =
            Kh + (size_t)(b * SEQ + jb + krow) * DMODEL + h * DK + kseg;
        stage16(&Kt[buf][krow][kseg], ks);
        stage16(&Kt[buf][krow][kseg + 8], ks + 8);
        const _Float16* vs =
            Vt + (size_t)(b * DMODEL + h * DK + vrow) * SEQ + jb + vseg;
        stage16(&Vl[buf][vrow][vseg], vs);
        stage16(&Vl[buf][vrow][vseg + 8], vs + 8);
    };

    stage_tiles(0, 0);   // prologue

    for (int jb = 0; jb < SEQ; jb += 32) {
        const int cur = (jb >> 5) & 1;

        // Wait for cur's staging (issued last iteration), make it visible.
        stage_wait();
        __syncthreads();

        // Kick off next block's staging; it completes during our compute.
        if (jb + 32 < SEQ) stage_tiles(cur ^ 1, jb + 32);

        // ---- S = Q K^T for 16x32 key block (two 16x16 tiles) ----
        v8f s0 = v8f{}, s1 = v8f{};
        s0 = wmma_f16(aq0, load_frag16(&Kt[cur][l16][0], hl), s0);
        s1 = wmma_f16(aq0, load_frag16(&Kt[cur][16 + l16][0], hl), s1);
        s0 = wmma_f16(aq1, load_frag16(&Kt[cur][l16][32], hl), s0);
        s1 = wmma_f16(aq1, load_frag16(&Kt[cur][16 + l16][32], hl), s1);

        // Key positions for this lane's two columns.
        const float* pk0 = pos + ((size_t)b * SEQ + jb + l16) * 3;
        const float* pk1 = pos + ((size_t)b * SEQ + jb + 16 + l16) * 3;
        const float k0x = pk0[0], k0y = pk0[1], k0z = pk0[2];
        const float k1x = pk1[0], k1y = pk1[1], k1z = pk1[2];

        float p0[8], p1[8];
#pragma unroll
        for (int e = 0; e < 8; ++e) {
            float dx = qx[e] - k0x, dy = qy[e] - k0y, dz = qz[e] - k0z;
            const float d0v = sqrtf(dx * dx + dy * dy + dz * dz);
            dx = qx[e] - k1x; dy = qy[e] - k1y; dz = qz[e] - k1z;
            const float d1v = sqrtf(dx * dx + dy * dy + dz * dz);
            const float a0 = s0[e] * scale + __logf(__expf(-ab * d0v) + 1e-8f);
            const float a1 = s1[e] * scale + __logf(__expf(-ab * d1v) + 1e-8f);

            // Row max across the 16 lanes sharing this row (xor within half).
            float v = fmaxf(a0, a1);
#pragma unroll
            for (int msk = 1; msk < 16; msk <<= 1)
                v = fmaxf(v, __shfl_xor(v, msk, 16));
            const float mn = fmaxf(m[e], v);
            const float sc = __expf(m[e] - mn);
            l[e] *= sc;
            o[0][e] *= sc; o[1][e] *= sc; o[2][e] *= sc; o[3][e] *= sc;

            p0[e] = __expf(a0 - mn);
            p1[e] = __expf(a1 - mn);
            float rs = p0[e] + p1[e];
#pragma unroll
            for (int msk = 1; msk < 16; msk <<= 1)
                rs += __shfl_xor(rs, msk, 16);
            l[e] += rs;
            m[e] = mn;
        }

        // ---- Relayout P (C/D layout -> A layout) through LDS ----
#pragma unroll
        for (int e = 0; e < 8; ++e) {
            const int lr = e + 8 * hl;
            Pt[wid][lr * 32 + l16]      = (_Float16)p0[e];
            Pt[wid][lr * 32 + 16 + l16] = (_Float16)p1[e];
        }
        __syncthreads();
        const v16h ap = load_frag16(&Pt[wid][l16 * 32], hl);

        // ---- O += P @ V (B-fragments from LDS V tile) ----
#pragma unroll
        for (int t = 0; t < 4; ++t) {
            const v16h bv = load_frag16(&Vl[cur][t * 16 + l16][0], hl);
            o[t] = wmma_f16(ap, bv, o[t]);
        }
        // No trailing barrier needed: next iteration's top barrier protects
        // the buffer being re-staged, and per-wave DS ops stay in order.
    }

    // Normalize and store per-head output into [B, N, DMODEL] f16.
#pragma unroll
    for (int t = 0; t < 4; ++t) {
#pragma unroll
        for (int e = 0; e < 8; ++e) {
            const int n = qbase + e + 8 * hl;
            const int c = h * DK + t * 16 + l16;
            O[(size_t)(b * SEQ + n) * DMODEL + c] = (_Float16)(o[t][e] / l[e]);
        }
    }
}

// ---------------------------------------------------------------------------
extern "C" void kernel_launch(void* const* d_in, const int* in_sizes, int n_in,
                              void* d_out, int out_size, void* d_ws, size_t ws_size,
                              hipStream_t stream) {
    const float* x   = (const float*)d_in[0];
    const float* pos = (const float*)d_in[1];
    const float* Wq  = (const float*)d_in[2];
    const float* bq  = (const float*)d_in[3];
    const float* Wk  = (const float*)d_in[4];
    const float* bk  = (const float*)d_in[5];
    const float* Wv  = (const float*)d_in[6];
    const float* bv  = (const float*)d_in[7];
    const float* Wo  = (const float*)d_in[8];
    const float* bo  = (const float*)d_in[9];
    const float* db  = (const float*)d_in[10];
    float* out = (float*)d_out;

    char* ws = (char*)d_ws;
    size_t off = 0;
    auto alloc = [&](size_t bytes) -> char* {
        char* p = ws + off;
        off += (bytes + 255) & ~(size_t)255;
        return p;
    };
    const size_t act_elems = (size_t)MROWS * DMODEL;     // 4M
    const size_t w_elems   = (size_t)DMODEL * DMODEL;    // 1M

    _Float16* xh  = (_Float16*)alloc(act_elems * 2);
    _Float16* wqh = (_Float16*)alloc(w_elems * 2);
    _Float16* wkh = (_Float16*)alloc(w_elems * 2);
    _Float16* wvh = (_Float16*)alloc(w_elems * 2);
    _Float16* woh = (_Float16*)alloc(w_elems * 2);
    _Float16* Qh  = (_Float16*)alloc(act_elems * 2);
    _Float16* Kh  = (_Float16*)alloc(act_elems * 2);
    _Float16* Vt  = (_Float16*)alloc(act_elems * 2);
    _Float16* Ah  = (_Float16*)alloc(act_elems * 2);
    (void)ws_size; (void)in_sizes; (void)n_in; (void)out_size;

    // f32 -> f16 conversions (~16 MB f16 total; trivially fast, L2 resident after).
    cvt_f32_f16<<<(int)((act_elems + 255) / 256), 256, 0, stream>>>(x,  xh,  (int)act_elems);
    cvt_f32_f16<<<(int)((w_elems + 255) / 256),   256, 0, stream>>>(Wq, wqh, (int)w_elems);
    cvt_f32_f16<<<(int)((w_elems + 255) / 256),   256, 0, stream>>>(Wk, wkh, (int)w_elems);
    cvt_f32_f16<<<(int)((w_elems + 255) / 256),   256, 0, stream>>>(Wv, wvh, (int)w_elems);
    cvt_f32_f16<<<(int)((w_elems + 255) / 256),   256, 0, stream>>>(Wo, woh, (int)w_elems);

    // QKV projections (V stored transposed for contiguous PV fragments).
    gemm16<<<512, 128, 0, stream>>>(xh, wqh, bq, Qh, nullptr, 0);
    gemm16<<<512, 128, 0, stream>>>(xh, wkh, bk, Kh, nullptr, 0);
    gemm16<<<512, 128, 0, stream>>>(xh, wvh, bv, Vt, nullptr, 1);

    // Fused flash attention with geometric bias.
    attn_fa<<<1024, 128, 0, stream>>>(Qh, Kh, Vt, pos, db, Ah);

    // Output projection -> f32 result.
    gemm16<<<512, 128, 0, stream>>>(Ah, woh, bo, nullptr, out, 2);
}
